// SNN_9749575762537
// MI455X (gfx1250) — matrix operand order repaired and checked by reading, not compiled, not run on hardware
//
#include <hip/hip_runtime.h>

typedef __attribute__((ext_vector_type(16))) _Float16 v16h;
typedef __attribute__((ext_vector_type(8)))  _Float16 v8h;
typedef __attribute__((ext_vector_type(8)))  float    v8f;

#define T_TOTAL 100
#define TCHUNK  25
#define NCHUNK  4
#define BATCH   128
#define NIN     1024
#define NH      2048
#define NOUT    512

// ---------------- fp32 -> fp16 conversion (8 elems / thread, vectorized) ---
__global__ void cvt_f32_to_f16_x8(const float* __restrict__ src,
                                  _Float16* __restrict__ dst, int n8) {
  int i = blockIdx.x * blockDim.x + threadIdx.x;
  if (i >= n8) return;
  const float4* s4 = (const float4*)src;
  float4 a = s4[2 * (size_t)i];
  float4 b = s4[2 * (size_t)i + 1];
  v8h h;
  h[0] = (_Float16)a.x; h[1] = (_Float16)a.y; h[2] = (_Float16)a.z; h[3] = (_Float16)a.w;
  h[4] = (_Float16)b.x; h[5] = (_Float16)b.y; h[6] = (_Float16)b.z; h[7] = (_Float16)b.w;
  *(v8h*)(dst + 8 * (size_t)i) = h;
}

__global__ void zero_f32(float* __restrict__ p, int n) {
  int i = blockIdx.x * blockDim.x + threadIdx.x;
  if (i < n) p[i] = 0.0f;
}

// ---------------- WMMA GEMM: C[M,N] = A[M,K] * W[N,K]^T  (f16 in, f32 out) --
// Wave tile: 16 rows x 64 cols, K-step 32 via v_wmma_f32_16x16x32_f16.
// Block = 8 waves covering 512 columns, grid = (M/16, N/512).
__global__ __launch_bounds__(256) void gemm_wmma_f16(
    const _Float16* __restrict__ A,   // M x K, row-major
    const _Float16* __restrict__ W,   // N x K, row-major  (== B col-major)
    float* __restrict__ C,            // M x N
    int M, int N, int K) {
  const int lane  = threadIdx.x & 31;
  const int wave  = threadIdx.x >> 5;
  const int mbase = blockIdx.x * 16;
  const int nbase = blockIdx.y * 512 + wave * 64;
  if (nbase >= N) return;                    // wave-uniform: EXEC stays all-1s

  const int hs  = lane >> 4;                 // half-wave select (0/1)
  const int l15 = lane & 15;

  // A-frag: 16 contiguous K halfs; lanes 0-15 take K [0..16), lanes 16-31 +16
  const _Float16* arow = A + (size_t)(mbase + l15) * K + (hs ? 16 : 0);
  // B-frag: lane = column, 16 contiguous K halfs; lanes 16-31 take K +16
  const _Float16* w0 = W + (size_t)(nbase +  0 + l15) * K + hs * 16;
  const _Float16* w1 = W + (size_t)(nbase + 16 + l15) * K + hs * 16;
  const _Float16* w2 = W + (size_t)(nbase + 32 + l15) * K + hs * 16;
  const _Float16* w3 = W + (size_t)(nbase + 48 + l15) * K + hs * 16;

  v8f acc0 = {}, acc1 = {}, acc2 = {}, acc3 = {};

  for (int k0 = 0; k0 < K; k0 += 32) {
    v8h alo = *(const v8h*)(arow + k0);
    v8h ahi = *(const v8h*)(arow + k0 + 8);
    v16h a;
#pragma unroll
    for (int j = 0; j < 8; ++j) { a[j] = alo[j]; a[8 + j] = ahi[j]; }
    v16h b0 = *(const v16h*)(w0 + k0);
    v16h b1 = *(const v16h*)(w1 + k0);
    v16h b2 = *(const v16h*)(w2 + k0);
    v16h b3 = *(const v16h*)(w3 + k0);
    acc0 = __builtin_amdgcn_wmma_f32_16x16x32_f16(false, a, false, b0, (short)0, acc0, false, false);
    acc1 = __builtin_amdgcn_wmma_f32_16x16x32_f16(false, a, false, b1, (short)0, acc1, false, false);
    acc2 = __builtin_amdgcn_wmma_f32_16x16x32_f16(false, a, false, b2, (short)0, acc2, false, false);
    acc3 = __builtin_amdgcn_wmma_f32_16x16x32_f16(false, a, false, b3, (short)0, acc3, false, false);
  }

  // C/D layout: VGPR r -> row r (lanes 0-15) / row 8+r (lanes 16-31)
#pragma unroll
  for (int r = 0; r < 8; ++r) {
    float* cp = C + (size_t)(mbase + r + hs * 8) * N + nbase + l15;
    cp[0]  = acc0[r];
    cp[16] = acc1[r];
    cp[32] = acc2[r];
    cp[48] = acc3[r];
  }
}

// ---------------- layer-1 leaky-integrate-fire scan over a T-chunk ---------
__global__ void scan1_kernel(const float* __restrict__ cur1,   // [Tc,B,NH]
                             const float* __restrict__ b1,
                             const float* __restrict__ bias1,
                             float* __restrict__ mem1,         // [B*NH] state
                             _Float16* __restrict__ spk1,      // [Tc,B,NH]
                             int Tc) {
  int i = blockIdx.x * blockDim.x + threadIdx.x;   // b*NH + h
  if (i >= BATCH * NH) return;
  const int h = i & (NH - 1);
  const float bias = b1[h] + bias1[0];
  float mem = mem1[i];
  for (int t = 0; t < Tc; ++t) {
    float cur = cur1[(size_t)t * (BATCH * NH) + i] + bias;
    float rst = (mem > 1.0f) ? 1.0f : 0.0f;        // reset from pre-update mem
    mem = 0.9f * mem + cur - rst;                  // BETA1, THR=1
    spk1[(size_t)t * (BATCH * NH) + i] = (mem > 1.0f) ? (_Float16)1.0f : (_Float16)0.0f;
  }
  mem1[i] = mem;
}

// ---------------- layer-2 scan, writes output spikes -----------------------
__global__ void scan2_kernel(const float* __restrict__ cur2,   // [Tc,B,NOUT]
                             const float* __restrict__ b2,
                             float* __restrict__ mem2,         // [B*NOUT] state
                             float* __restrict__ out,          // [Tc,B,NOUT]
                             int Tc) {
  int i = blockIdx.x * blockDim.x + threadIdx.x;   // b*NOUT + n
  if (i >= BATCH * NOUT) return;
  const int n = i & (NOUT - 1);
  const float bb = b2[n];
  float mem = mem2[i];
  for (int t = 0; t < Tc; ++t) {
    float cur = cur2[(size_t)t * (BATCH * NOUT) + i] + bb;
    float rst = (mem > 1.0f) ? 1.0f : 0.0f;
    mem = 0.8f * mem + cur - rst;                  // BETA2
    out[(size_t)t * (BATCH * NOUT) + i] = (mem > 1.0f) ? 1.0f : 0.0f;
  }
  mem2[i] = mem;
}

extern "C" void kernel_launch(void* const* d_in, const int* in_sizes, int n_in,
                              void* d_out, int out_size, void* d_ws, size_t ws_size,
                              hipStream_t stream) {
  const float* x     = (const float*)d_in[0];  // [T,B,NIN]
  const float* W1    = (const float*)d_in[1];  // [NH,NIN]
  const float* b1    = (const float*)d_in[2];  // [NH]
  const float* bias1 = (const float*)d_in[3];  // [1]
  const float* W2    = (const float*)d_in[4];  // [NOUT,NH]
  const float* b2    = (const float*)d_in[5];  // [NOUT]
  float* out = (float*)d_out;

  // ---- workspace carve-out (~60 MB total) ----
  char* ws = (char*)d_ws;
  _Float16* W1h = (_Float16*)ws;  ws += sizeof(_Float16) * (size_t)NH * NIN;
  _Float16* W2h = (_Float16*)ws;  ws += sizeof(_Float16) * (size_t)NOUT * NH;
  float*    mem = (float*)ws;     ws += sizeof(float) * (size_t)(BATCH * NH + BATCH * NOUT);
  float*    mem1 = mem;
  float*    mem2 = mem + BATCH * NH;
  _Float16* xh   = (_Float16*)ws; ws += sizeof(_Float16) * (size_t)TCHUNK * BATCH * NIN;
  float*    cur1 = (float*)ws;    ws += sizeof(float)    * (size_t)TCHUNK * BATCH * NH;
  _Float16* spk1 = (_Float16*)ws; ws += sizeof(_Float16) * (size_t)TCHUNK * BATCH * NH;
  float*    cur2 = (float*)ws;    ws += sizeof(float)    * (size_t)TCHUNK * BATCH * NOUT;

  // weights -> f16 (every call: deterministic, capture-safe)
  {
    int n8 = (NH * NIN) / 8;
    cvt_f32_to_f16_x8<<<(n8 + 255) / 256, 256, 0, stream>>>(W1, W1h, n8);
    n8 = (NOUT * NH) / 8;
    cvt_f32_to_f16_x8<<<(n8 + 255) / 256, 256, 0, stream>>>(W2, W2h, n8);
    int nm = BATCH * NH + BATCH * NOUT;
    zero_f32<<<(nm + 255) / 256, 256, 0, stream>>>(mem, nm);
  }

  const int MC = TCHUNK * BATCH;               // 3200 rows per chunk
  for (int c = 0; c < NCHUNK; ++c) {
    const float* xc = x + (size_t)c * TCHUNK * BATCH * NIN;

    int n8 = (TCHUNK * BATCH * NIN) / 8;
    cvt_f32_to_f16_x8<<<(n8 + 255) / 256, 256, 0, stream>>>(xc, xh, n8);

    dim3 g1(MC / 16, NH / 512);
    gemm_wmma_f16<<<g1, 256, 0, stream>>>(xh, W1h, cur1, MC, NH, NIN);

    scan1_kernel<<<(BATCH * NH + 255) / 256, 256, 0, stream>>>(
        cur1, b1, bias1, mem1, spk1, TCHUNK);

    dim3 g2(MC / 16, NOUT / 512);
    gemm_wmma_f16<<<g2, 256, 0, stream>>>(spk1, W2h, cur2, MC, NOUT, NH);

    scan2_kernel<<<(BATCH * NOUT + 255) / 256, 256, 0, stream>>>(
        cur2, b2, mem2, out + (size_t)c * TCHUNK * BATCH * NOUT, TCHUNK);
  }
}
